// CustomMultiheadAttention_42606075576798
// MI455X (gfx1250) — matrix operand order repaired and checked by reading, not compiled, not run on hardware
//
#include <hip/hip_runtime.h>
#include <hip/hip_bf16.h>

#define D_MODEL  1024
#define N_HEAD   16
#define HEAD_DIM 64
#define BATCH    2
#define SEQ      2048
#define M_TOK    (BATCH * SEQ)   // 4096 tokens

typedef __bf16 v16bf __attribute__((ext_vector_type(16)));
typedef __bf16 v2bf  __attribute__((ext_vector_type(2)));
typedef float  v8f   __attribute__((ext_vector_type(8)));

union AFrag { v16bf v; unsigned short h[16]; unsigned int u[8]; uint4 q[2]; };
union CFrag { v8f v; float f[8]; };

// Native fp32->bf16 conversions (lower to v_cvt_pk_bf16_f32 on gfx1250).
__device__ __forceinline__ unsigned short f32_to_bf16_rne(float x) {
  __bf16 b = (__bf16)x;
  return __builtin_bit_cast(unsigned short, b);
}
__device__ __forceinline__ unsigned pack_bf16x2(float lo, float hi) {
  v2bf p = { (__bf16)lo, (__bf16)hi };
  return __builtin_bit_cast(unsigned, p);
}

// Raw LDS byte offset of a shared-memory pointer (addrspace(3) ptrtoint).
__device__ __forceinline__ unsigned lds_off(const void* p) {
  return (unsigned)(unsigned long long)(__attribute__((address_space(3))) const void*)p;
}

// Async global->LDS 16-byte copy (per active lane). Tracked by ASYNCcnt.
__device__ __forceinline__ void async_load_b128(unsigned lds_addr,
                                                const void* gaddr) {
  asm volatile("global_load_async_to_lds_b128 %0, %1, off"
               :: "v"(lds_addr), "v"((unsigned long long)(size_t)gaddr)
               : "memory");
}
__device__ __forceinline__ void wait_async0() {
  asm volatile("s_wait_asynccnt 0x0" ::: "memory");
}

// ---------------------------------------------------------------------------
// GEMM: C[M,N] = A[M,K] @ W[K,N] + bias, bf16 WMMA, fp32 accum.
// MODE 0: fp32 C row-major (final projection).
// MODE 1: bf16 into [B, H, L, 64] head layout (Q, K), scaled by outScale.
// MODE 2: bf16 into [B, H, 64, L] transposed head layout (V), b128 stores.
// Tile 128x128x32, 256 threads = 8 waves, wave sub-tile 64x32 (4x2 frags).
// ---------------------------------------------------------------------------
template <int MODE>
__global__ void __launch_bounds__(256)
gemm_bf16(const float* __restrict__ A, const float* __restrict__ W,
          const float* __restrict__ bias, float* __restrict__ Cout,
          unsigned short* __restrict__ Hout, float outScale) {
  const int K = D_MODEL, N = D_MODEL;
  __shared__ unsigned short As[128 * 32];   // [m][k] bf16
  __shared__ unsigned short Bs[128 * 32];   // [n][k] bf16 (transposed for B-frag)

  const int tid  = threadIdx.x;
  const int lane = tid & 31;
  const int wave = tid >> 5;
  const int wm = (wave >> 2) * 64;          // wave row base in tile
  const int wn = (wave & 3) * 32;           // wave col base in tile
  const int m0 = blockIdx.y * 128;
  const int n0 = blockIdx.x * 128;

  CFrag acc[4][2] = {};

  const int arow = lane & 15;
  const int akb  = (lane >> 4) << 3;        // A-frag k offset: 0 or 8
  const int bkb  = (lane >> 4) << 3;        // B-frag k offset: 0 or 8

  for (int kt = 0; kt < K; kt += 32) {
    // prefetch next k-tile of the streaming operands
    if (kt + 32 < K) {
      int prow = tid >> 1;                  // 2 threads cover a 128B A row seg
      __builtin_prefetch(A + (size_t)(m0 + prow) * K + kt + 32 +
                         ((tid & 1) << 4), 0, 0);
      int pkr = tid >> 3;                   // 8 threads cover a 512B W row
      __builtin_prefetch(W + (size_t)(kt + 32 + pkr) * N + n0 +
                         ((tid & 7) << 4), 0, 0);
    }
    // --- stage A tile (128x32 fp32 -> bf16) ---
    #pragma unroll
    for (int p = 0; p < 4; ++p) {
      int idx = tid + p * 256;              // 0..1023 float4 slots
      int row = idx >> 3;
      int kq  = (idx & 7) << 2;
      const float4 v4 = *(const float4*)(A + (size_t)(m0 + row) * K + kt + kq);
      *(unsigned*)(&As[row * 32 + kq])     = pack_bf16x2(v4.x, v4.y);
      *(unsigned*)(&As[row * 32 + kq + 2]) = pack_bf16x2(v4.z, v4.w);
    }
    // --- stage W tile (32x128 fp32 -> bf16, transposed into Bs[n][k]) ---
    #pragma unroll
    for (int p = 0; p < 4; ++p) {
      int idx = tid + p * 256;
      int kr  = idx >> 5;                   // 0..31
      int nq  = (idx & 31) << 2;
      const float4 v4 = *(const float4*)(W + (size_t)(kt + kr) * N + n0 + nq);
      Bs[(nq + 0) * 32 + kr] = f32_to_bf16_rne(v4.x);
      Bs[(nq + 1) * 32 + kr] = f32_to_bf16_rne(v4.y);
      Bs[(nq + 2) * 32 + kr] = f32_to_bf16_rne(v4.z);
      Bs[(nq + 3) * 32 + kr] = f32_to_bf16_rne(v4.w);
    }
    __syncthreads();

    // --- A fragments: ISA 16-bit A layout == two contiguous 16B runs ---
    AFrag af[4];
    #pragma unroll
    for (int i = 0; i < 4; ++i) {
      const unsigned short* ap = &As[(wm + i * 16 + arow) * 32];
      af[i].q[0] = *(const uint4*)(ap + akb);
      af[i].q[1] = *(const uint4*)(ap + akb + 16);
    }
    // --- B fragments (lane holds col n; k 8h..8h+7 and 16+8h..16+8h+7) ---
    AFrag bf[2];
    #pragma unroll
    for (int j = 0; j < 2; ++j) {
      const unsigned short* bp = &Bs[(wn + j * 16 + (lane & 15)) * 32 + bkb];
      bf[j].q[0] = *(const uint4*)(bp);
      bf[j].q[1] = *(const uint4*)(bp + 16);
    }
    #pragma unroll
    for (int i = 0; i < 4; ++i)
      #pragma unroll
      for (int j = 0; j < 2; ++j)
        acc[i][j].v = __builtin_amdgcn_wmma_f32_16x16x32_bf16(
            false, af[i].v, false, bf[j].v, (short)0, acc[i][j].v, false, false);

    __syncthreads();
  }

  // --- epilogue ---
  #pragma unroll
  for (int i = 0; i < 4; ++i) {
    int mbase = m0 + wm + i * 16 + ((lane >> 4) << 3);   // 8 consecutive rows
    #pragma unroll
    for (int j = 0; j < 2; ++j) {
      int col = n0 + wn + j * 16 + (lane & 15);
      float bv = bias[col];
      if (MODE == 2) {
        // V^T layout: lane's 8 rows are 8 consecutive l -> one b128 store
        int b = mbase >> 11;                // / SEQ
        int l0 = mbase & (SEQ - 1);
        int h = col >> 6;
        int d = col & 63;
        unsigned short* dst =
            Hout + (((size_t)(b * N_HEAD + h)) * HEAD_DIM + d) * SEQ + l0;
        uint4 q;
        q.x = pack_bf16x2((acc[i][j].f[0] + bv) * outScale,
                          (acc[i][j].f[1] + bv) * outScale);
        q.y = pack_bf16x2((acc[i][j].f[2] + bv) * outScale,
                          (acc[i][j].f[3] + bv) * outScale);
        q.z = pack_bf16x2((acc[i][j].f[4] + bv) * outScale,
                          (acc[i][j].f[5] + bv) * outScale);
        q.w = pack_bf16x2((acc[i][j].f[6] + bv) * outScale,
                          (acc[i][j].f[7] + bv) * outScale);
        *(uint4*)dst = q;
      } else {
        #pragma unroll
        for (int r = 0; r < 8; ++r) {
          int m = mbase + r;
          float val = (acc[i][j].f[r] + bv) * outScale;
          if (MODE == 0) {
            Cout[(size_t)m * N + col] = val;
          } else {                          // MODE 1: [B,H,L,64]
            int b = m >> 11;
            int l = m & (SEQ - 1);
            int h = col >> 6;
            int d = col & 63;
            Hout[(((size_t)(b * N_HEAD + h)) * SEQ + l) * HEAD_DIM + d] =
                f32_to_bf16_rne(val);
          }
        }
      }
    }
  }
}

// ---------------------------------------------------------------------------
// Flash attention: grid (L/64, B*H), 128 threads (4 waves). Each wave owns 16
// query rows; loop over 64-row KV tiles with online softmax. K is [B,H,L,64],
// V is pre-transposed [B,H,64,L]; both tiles are staged into double-buffered
// LDS with async global->LDS b128 copies overlapping the current tile's math.
// Scale 1/sqrt(64) was folded into Q.
// ---------------------------------------------------------------------------
__global__ void __launch_bounds__(128)
flash_attn(const unsigned short* __restrict__ Qb,
           const unsigned short* __restrict__ Kb,
           const unsigned short* __restrict__ Vb,
           float* __restrict__ Oout) {
  __shared__ unsigned short Ks[2][64 * 64];     // [s][d]
  __shared__ unsigned short Vt[2][64 * 64];     // [d][s]
  __shared__ unsigned short Ps[4][16 * 64];     // per-wave P scratch [row][kv]

  const int tid  = threadIdx.x;
  const int lane = tid & 31;
  const int wave = tid >> 5;
  const int bh = blockIdx.y;                    // b*H + h
  const int q0 = blockIdx.x * 64 + wave * 16;
  const size_t headBase = (size_t)bh * SEQ * HEAD_DIM;

  const int akb = (lane >> 4) << 3;             // A-frag k offset
  const int bkb = (lane >> 4) << 3;             // B-frag k offset

  const unsigned ksBase = lds_off(&Ks[0][0]);
  const unsigned vtBase = lds_off(&Vt[0][0]);

  // issue async stage of KV tile t into buffer buf
  auto stage = [&](int t, int buf) {
    const unsigned short* kg = Kb + headBase + (size_t)t * 64 * HEAD_DIM;
    const unsigned short* vg = Vb + headBase + (size_t)t * 64;   // col t*64
    unsigned kb = ksBase + buf * 8192;
    unsigned vb = vtBase + buf * 8192;
    #pragma unroll
    for (int p = 0; p < 4; ++p) {
      int idx = tid + p * 128;                  // 0..511 b128 slots
      async_load_b128(kb + idx * 16, kg + idx * 8);
      int row = idx >> 3;                       // d row (0..63)
      int c8  = (idx & 7) * 8;                  // s offset within tile
      async_load_b128(vb + (row * 64 + c8) * 2, vg + (size_t)row * SEQ + c8);
    }
  };

  // Q fragments: 16 rows x 64 (two k-chunks of 32), two b128 runs per chunk
  AFrag qf[2];
  {
    const unsigned short* qp =
        Qb + headBase + (size_t)(q0 + (lane & 15)) * HEAD_DIM;
    #pragma unroll
    for (int c = 0; c < 2; ++c) {
      qf[c].q[0] = *(const uint4*)(qp + 32 * c + akb);
      qf[c].q[1] = *(const uint4*)(qp + 32 * c + akb + 16);
    }
  }

  CFrag o[4] = {};                              // 16x64 fp32 accum
  float mrow[8], lrow[8];
  #pragma unroll
  for (int r = 0; r < 8; ++r) { mrow[r] = -1e30f; lrow[r] = 0.f; }

  stage(0, 0);
  wait_async0();
  __syncthreads();

  for (int t = 0; t < SEQ / 64; ++t) {
    const int buf = t & 1;
    if (t + 1 < SEQ / 64) stage(t + 1, buf ^ 1);   // overlap copy with math

    const unsigned short* KsT = Ks[buf];
    const unsigned short* VtT = Vt[buf];

    // S = Q @ K^T : 4 column sub-tiles of 16 kv positions
    CFrag sfr[4] = {};
    #pragma unroll
    for (int n = 0; n < 4; ++n) {
      #pragma unroll
      for (int c = 0; c < 2; ++c) {
        AFrag bfrag;
        const unsigned short* bp =
            &KsT[(n * 16 + (lane & 15)) * 64 + 32 * c + bkb];
        bfrag.q[0] = *(const uint4*)bp;
        bfrag.q[1] = *(const uint4*)(bp + 16);
        sfr[n].v = __builtin_amdgcn_wmma_f32_16x16x32_bf16(
            false, qf[c].v, false, bfrag.v, (short)0, sfr[n].v, false, false);
      }
    }

    // online softmax (row reductions across each 16-lane half)
    float newmax[8], alpha[8], rs[8];
    #pragma unroll
    for (int r = 0; r < 8; ++r) {
      float mx = fmaxf(fmaxf(sfr[0].f[r], sfr[1].f[r]),
                       fmaxf(sfr[2].f[r], sfr[3].f[r]));
      #pragma unroll
      for (int sh = 1; sh < 16; sh <<= 1)
        mx = fmaxf(mx, __shfl_xor(mx, sh, 32));
      newmax[r] = fmaxf(mrow[r], mx);
      alpha[r] = __expf(mrow[r] - newmax[r]);
      rs[r] = 0.f;
    }
    #pragma unroll
    for (int n = 0; n < 4; ++n)
      #pragma unroll
      for (int r = 0; r < 8; ++r) {
        float p = __expf(sfr[n].f[r] - newmax[r]);
        sfr[n].f[r] = p;
        rs[r] += p;
      }
    #pragma unroll
    for (int r = 0; r < 8; ++r) {
      #pragma unroll
      for (int sh = 1; sh < 16; sh <<= 1)
        rs[r] += __shfl_xor(rs[r], sh, 32);
      lrow[r] = lrow[r] * alpha[r] + rs[r];
      mrow[r] = newmax[r];
    }
    #pragma unroll
    for (int n = 0; n < 4; ++n)
      #pragma unroll
      for (int r = 0; r < 8; ++r)
        o[n].f[r] *= alpha[r];

    // P: C-frag layout -> per-wave LDS [row][kv] as bf16
    {
      unsigned short* pw = Ps[wave];
      int rbase = (lane >> 4) << 3;
      #pragma unroll
      for (int n = 0; n < 4; ++n) {
        int col = n * 16 + (lane & 15);
        #pragma unroll
        for (int r = 0; r < 8; ++r)
          pw[(rbase + r) * 64 + col] = f32_to_bf16_rne(sfr[n].f[r]);
      }
    }

    // O += P @ V
    {
      AFrag pf[2];
      const unsigned short* pp = Ps[wave] + (size_t)(lane & 15) * 64;
      #pragma unroll
      for (int c = 0; c < 2; ++c) {
        pf[c].q[0] = *(const uint4*)(pp + 32 * c + akb);
        pf[c].q[1] = *(const uint4*)(pp + 32 * c + akb + 16);
      }
      #pragma unroll
      for (int n = 0; n < 4; ++n)
        #pragma unroll
        for (int c = 0; c < 2; ++c) {
          AFrag vfrag;
          const unsigned short* vp =
              &VtT[(n * 16 + (lane & 15)) * 64 + 32 * c + bkb];
          vfrag.q[0] = *(const uint4*)vp;
          vfrag.q[1] = *(const uint4*)(vp + 16);
          o[n].v = __builtin_amdgcn_wmma_f32_16x16x32_bf16(
              false, pf[c].v, false, vfrag.v, (short)0, o[n].v, false, false);
        }
    }

    wait_async0();        // next tile's async copy has landed (own wave)
    __syncthreads();      // all waves done reading buf & done staging buf^1
  }

  // normalize rows and store to [B*L, D_MODEL] attention output
  const int b = bh >> 4;
  const int h = bh & 15;
  #pragma unroll
  for (int r = 0; r < 8; ++r) {
    float inv = 1.f / lrow[r];
    int row = q0 + ((lane >> 4) << 3) + r;
    float* op = Oout + ((size_t)(b * SEQ + row)) * D_MODEL + h * HEAD_DIM;
    #pragma unroll
    for (int n = 0; n < 4; ++n)
      op[n * 16 + (lane & 15)] = o[n].f[r] * inv;
  }
}

// ---------------------------------------------------------------------------
extern "C" void kernel_launch(void* const* d_in, const int* in_sizes, int n_in,
                              void* d_out, int out_size, void* d_ws,
                              size_t ws_size, hipStream_t stream) {
  (void)in_sizes; (void)n_in; (void)out_size; (void)ws_size;
  const float* x  = (const float*)d_in[0];
  const float* Wq = (const float*)d_in[1];
  const float* bq = (const float*)d_in[2];
  const float* Wk = (const float*)d_in[3];
  const float* bk = (const float*)d_in[4];
  const float* Wv = (const float*)d_in[5];
  const float* bv = (const float*)d_in[6];
  const float* Wo = (const float*)d_in[7];
  const float* bo = (const float*)d_in[8];
  float* out = (float*)d_out;

  const size_t qkvElems = (size_t)BATCH * N_HEAD * SEQ * HEAD_DIM;
  unsigned short* Qb = (unsigned short*)d_ws;
  unsigned short* Kb = Qb + qkvElems;
  unsigned short* Vb = Kb + qkvElems;
  float* attn = (float*)(Vb + qkvElems);

  dim3 gg(D_MODEL / 128, M_TOK / 128);          // (8, 32)
  // QKV projections (bf16 WMMA). Softmax scale 1/sqrt(64) folded into Q.
  gemm_bf16<1><<<gg, 256, 0, stream>>>(x, Wq, bq, nullptr, Qb, 0.125f);
  gemm_bf16<1><<<gg, 256, 0, stream>>>(x, Wk, bk, nullptr, Kb, 1.0f);
  gemm_bf16<2><<<gg, 256, 0, stream>>>(x, Wv, bv, nullptr, Vb, 1.0f);  // V^T
  // Flash attention over 64-row KV tiles (async double-buffered staging).
  flash_attn<<<dim3(SEQ / 64, BATCH * N_HEAD), 128, 0, stream>>>(Qb, Kb, Vb,
                                                                 attn);
  // Output projection (fp32 result + bias).
  gemm_bf16<0><<<gg, 256, 0, stream>>>(attn, Wo, bo, out, nullptr, 1.0f);
}
